// RecursiveWeight_11252814316096
// MI455X (gfx1250) — matrix operand order repaired and checked by reading, not compiled, not run on hardware
//
#include <hip/hip_runtime.h>

typedef __attribute__((ext_vector_type(16))) __bf16 v16bf;
typedef __attribute__((ext_vector_type(8)))  __bf16 v8bf;
typedef __attribute__((ext_vector_type(8)))  float  v8f;

#if __has_builtin(__builtin_amdgcn_sched_barrier)
#define SCHED_FENCE() __builtin_amdgcn_sched_barrier(0)
#else
#define SCHED_FENCE()
#endif

constexpr int   Vv      = 200000;
constexpr int   D       = 512;
constexpr int   Nn      = 32768;
constexpr int   R       = 8;
constexpr int   H       = 16;
constexpr float CLAMP_V = 1000.0f;

constexpr int ROWS_PER_WG = 128;           // output rows per workgroup
constexpr int KC          = 32;            // bf16 WMMA K per step
constexpr int NKC         = (R * D) / KC;  // 128 k-chunks
constexpr int LDA         = 40;            // padded LDS stride (bf16) for A
constexpr int LDB         = 40;            // padded LDS stride (bf16) for B

__device__ __forceinline__ unsigned short f32_to_bf16(float f) {
  unsigned u = __float_as_uint(f);
  unsigned r = (u + 0x7FFFu + ((u >> 16) & 1u)) >> 16;  // round-to-nearest-even
  return (unsigned short)r;
}

// ---------------------------------------------------------------------------
// Prep 1: Tt[r][e][d] = bf16( transform[r][d][e] * contrib_w[r] )
// ---------------------------------------------------------------------------
__global__ void prep_transform(const float* __restrict__ transform,
                               const float* __restrict__ contrib_w,
                               unsigned short* __restrict__ Tt) {
  __shared__ float tile[32][33];
  const int r  = blockIdx.z;
  const int eb = blockIdx.x * 32;
  const int db = blockIdx.y * 32;
  const float w = contrib_w[r];
  const int x = threadIdx.x;   // 0..31
  const int y = threadIdx.y;   // 0..7
  const float* src = transform + (size_t)r * D * D;
#pragma unroll
  for (int i = 0; i < 4; ++i) {
    int d = db + y + i * 8;
    tile[y + i * 8][x] = src[(size_t)d * D + (eb + x)];
  }
  __syncthreads();
#pragma unroll
  for (int i = 0; i < 4; ++i) {
    int e = eb + y + i * 8;
    Tt[((size_t)r * D + e) * D + (db + x)] = f32_to_bf16(tile[x][y + i * 8] * w);
  }
}

// ---------------------------------------------------------------------------
// Prep 2: P[e] = clip(base_phase[e] + sum_h amp[h][e]*sin(f_h*t + o_h)) + err[e]
// ---------------------------------------------------------------------------
__global__ void prep_phase(const float* __restrict__ base_phase,
                           const float* __restrict__ amp,
                           const float* __restrict__ freq,
                           const float* __restrict__ off,
                           const float* __restrict__ err,
                           const int*   __restrict__ tstep,
                           float* __restrict__ P) {
  const int e = threadIdx.x;
  const float t = (float)tstep[0];
  float s = base_phase[e];
#pragma unroll
  for (int h = 0; h < H; ++h)
    s += amp[h * D + e] * __sinf(freq[h] * t + off[h]);
  s = fminf(fmaxf(s, -CLAMP_V), CLAMP_V);
  P[e] = s + err[e];
}

// ---------------------------------------------------------------------------
// Main: gather-GEMM via v_wmma_f32_16x16x32_bf16.
// Register-staged software pipeline (globals for kc+1 in flight while kc
// computes from LDS; commit after the post-compute barrier -> single LDS
// buffer). B fragments double-buffered in registers so ds waits overlap WMMA.
// ---------------------------------------------------------------------------
__global__ __launch_bounds__(1024) void gather_gemm(
    const float* __restrict__ codebook,
    const int* __restrict__ base_idx,
    const int* __restrict__ ref_idx,
    const unsigned short* __restrict__ Tt,
    const float* __restrict__ P,
    float* __restrict__ out) {
  __shared__ __align__(16) unsigned short As[ROWS_PER_WG * LDA];  // 128x32 bf16
  __shared__ __align__(16) unsigned short Bs[D * LDB];            // 512x32 bf16
  __shared__ int idxs[ROWS_PER_WG * R];

  const int t    = threadIdx.x;
  const int lane = t & 31;
  const int wave = t >> 5;        // 0..31
  const int wrow = wave & 7;      // row tile within WG
  const int wcg  = wave >> 3;     // column group 0..3
  const int n0   = blockIdx.x * ROWS_PER_WG;

  // Loader-role coordinates (uniform per thread across iterations)
  const int arow = t >> 3, aq = t & 7;   // A: 8 threads/row, 16B each
  const int be0  = t >> 2, bq = t & 3;   // B: 4 threads/e-row, 16B each (+256)

  // Preload all ref indices for this row block (coalesced).
  idxs[t] = ref_idx[(size_t)n0 * R + t];
  __syncthreads();

  v8f acc[8];
#pragma unroll
  for (int i = 0; i < 8; ++i) acc[i] = (v8f)(0.0f);

  const int m  = lane & 15;   // A row / B column within tile
  const int kh = lane >> 4;   // K-half select (fragment layout)

  // ---- pipeline stage registers
  float4 aReg;
  uint4  bReg0, bReg1;

  // prologue: fetch + commit chunk 0
  {
    const int ci = idxs[arow * R + 0];
    aReg  = reinterpret_cast<const float4*>(codebook + (size_t)ci * D)[aq];
    bReg0 = *reinterpret_cast<const uint4*>(Tt + (size_t)be0 * D + bq * 8);
    bReg1 = *reinterpret_cast<const uint4*>(Tt + ((size_t)be0 + 256) * D + bq * 8);
    uint2 pk;
    pk.x = (unsigned)f32_to_bf16(aReg.x) | ((unsigned)f32_to_bf16(aReg.y) << 16);
    pk.y = (unsigned)f32_to_bf16(aReg.z) | ((unsigned)f32_to_bf16(aReg.w) << 16);
    *reinterpret_cast<uint2*>(&As[arow * LDA + aq * 4]) = pk;
    *reinterpret_cast<uint4*>(&Bs[be0 * LDB + bq * 8])         = bReg0;
    *reinterpret_cast<uint4*>(&Bs[(be0 + 256) * LDB + bq * 8]) = bReg1;
  }
  __syncthreads();

  for (int kc = 0; kc < NKC; ++kc) {
    // ---- issue global loads for chunk kc+1 (land during compute below)
    if (kc + 1 < NKC) {
      const int r1  = (kc + 1) >> 4;
      const int d01 = ((kc + 1) & 15) * KC;
      const int ci  = idxs[arow * R + r1];
      aReg  = reinterpret_cast<const float4*>(codebook + (size_t)ci * D + d01)[aq];
      bReg0 = *reinterpret_cast<const uint4*>(
          Tt + ((size_t)r1 * D + be0) * D + d01 + bq * 8);
      bReg1 = *reinterpret_cast<const uint4*>(
          Tt + ((size_t)r1 * D + be0 + 256) * D + d01 + bq * 8);
    }
    // ---- warm L2 for the kc+2 codebook gather (HBM critical path)
    if (kc + 2 < NKC) {
      const int r2  = (kc + 2) >> 4;
      const int d02 = ((kc + 2) & 15) * KC;
      const int ci2 = idxs[arow * R + r2];
      __builtin_prefetch(codebook + (size_t)ci2 * D + d02 + aq * 4, 0, 0);
    }
    SCHED_FENCE();  // staged loads issued; nothing below may hoist above

    // ---- compute chunk kc from LDS
    const int rowl = wrow * 16 + m;
    const v8bf alo = *reinterpret_cast<const v8bf*>(&As[rowl * LDA + kh * 16]);
    const v8bf ahi = *reinterpret_cast<const v8bf*>(&As[rowl * LDA + kh * 16 + 8]);
    v16bf a;
#pragma unroll
    for (int i = 0; i < 8; ++i) { a[i] = alo[i]; a[8 + i] = ahi[i]; }

    // B fragments double-buffered: load tile tt+1 while WMMA tt executes.
    const int ebase = wcg * 128 + m;
    v8bf blo = *reinterpret_cast<const v8bf*>(&Bs[ebase * LDB + kh * 16]);
    v8bf bhi = *reinterpret_cast<const v8bf*>(&Bs[ebase * LDB + kh * 16 + 8]);
#pragma unroll
    for (int tt = 0; tt < 8; ++tt) {
      v8bf nlo = blo, nhi = bhi;
      if (tt < 7) {
        const int e = ebase + (tt + 1) * 16;
        nlo = *reinterpret_cast<const v8bf*>(&Bs[e * LDB + kh * 16]);
        nhi = *reinterpret_cast<const v8bf*>(&Bs[e * LDB + kh * 16 + 8]);
      }
      v16bf b;
#pragma unroll
      for (int i = 0; i < 8; ++i) { b[i] = blo[i]; b[8 + i] = bhi[i]; }
      acc[tt] = __builtin_amdgcn_wmma_f32_16x16x32_bf16(
          false, a, false, b, (short)0, acc[tt], false, false);
      blo = nlo; bhi = nhi;
    }
    SCHED_FENCE();  // keep staged-register consumers (bf16 pack) below WMMAs

    __syncthreads();  // everyone done reading As/Bs for chunk kc

    // ---- commit chunk kc+1 to LDS (single buffer: safe after the barrier)
    if (kc + 1 < NKC) {
      uint2 pk;
      pk.x = (unsigned)f32_to_bf16(aReg.x) | ((unsigned)f32_to_bf16(aReg.y) << 16);
      pk.y = (unsigned)f32_to_bf16(aReg.z) | ((unsigned)f32_to_bf16(aReg.w) << 16);
      *reinterpret_cast<uint2*>(&As[arow * LDA + aq * 4]) = pk;
      *reinterpret_cast<uint4*>(&Bs[be0 * LDB + bq * 8])         = bReg0;
      *reinterpret_cast<uint4*>(&Bs[(be0 + 256) * LDB + bq * 8]) = bReg1;
    }
    __syncthreads();  // chunk kc+1 visible for next iteration
  }

  // ---- Epilogue: out = clip(contrib + codebook[base_idx] + P, +-1000)
#pragma unroll
  for (int tt = 0; tt < 8; ++tt) {
    const int col = wcg * 128 + tt * 16 + (lane & 15);
    const float pc = P[col];
#pragma unroll
    for (int v = 0; v < 8; ++v) {
      const int row = n0 + wrow * 16 + (v + 8 * (lane >> 4));  // C/D layout
      const int bi  = base_idx[row];
      float val = acc[tt][v] + codebook[(size_t)bi * D + col] + pc;
      val = fminf(fmaxf(val, -CLAMP_V), CLAMP_V);
      out[(size_t)row * D + col] = val;
    }
  }
}

// ---------------------------------------------------------------------------
extern "C" void kernel_launch(void* const* d_in, const int* in_sizes, int n_in,
                              void* d_out, int out_size, void* d_ws, size_t ws_size,
                              hipStream_t stream) {
  const float* codebook   = (const float*)d_in[0];
  const int*   base_idx   = (const int*)d_in[1];
  const int*   ref_idx    = (const int*)d_in[2];
  const float* transform  = (const float*)d_in[3];
  const float* contrib_w  = (const float*)d_in[4];
  const float* base_phase = (const float*)d_in[5];
  const float* harm_amp   = (const float*)d_in[6];
  const float* freqs      = (const float*)d_in[7];
  const float* phase_off  = (const float*)d_in[8];
  const float* err        = (const float*)d_in[9];
  const int*   time_step  = (const int*)d_in[10];
  float* out = (float*)d_out;

  // Workspace: Tt (R*D*D bf16 = 4 MB) | P (D f32)
  unsigned short* Tt = (unsigned short*)d_ws;
  float* P = (float*)((char*)d_ws + (size_t)R * D * D * sizeof(unsigned short));

  dim3 tb(32, 8, 1);
  dim3 tg(D / 32, D / 32, R);
  prep_transform<<<tg, tb, 0, stream>>>(transform, contrib_w, Tt);
  prep_phase<<<1, D, 0, stream>>>(base_phase, harm_amp, freqs, phase_off, err,
                                  time_step, P);
  gather_gemm<<<Nn / ROWS_PER_WG, 1024, 0, stream>>>(codebook, base_idx,
                                                     ref_idx, Tt, P, out);
}
